// KVCachedMultiHeadAttention_75737453297744
// MI455X (gfx1250) — compile-verified, matrix operand order checked by reading, not run on hardware
//
#include <hip/hip_runtime.h>

#define DEV __device__ __forceinline__

typedef __attribute__((ext_vector_type(16))) __bf16           bf16x16;
typedef __attribute__((ext_vector_type(8)))  float            floatx8;
typedef __attribute__((ext_vector_type(8)))  unsigned short   ushort8;
typedef __attribute__((ext_vector_type(16))) unsigned short   ushort16;
typedef __attribute__((ext_vector_type(4)))  int              int4v;

union FragCvt { ushort16 u; bf16x16 v; };

struct US4 { unsigned short a, b, c, d; };

#ifndef __has_builtin
#define __has_builtin(x) 0
#endif
#if __has_builtin(__builtin_amdgcn_global_load_async_to_lds_b128)
#define HAVE_ASYNC 1
#else
#define HAVE_ASYNC 0
#endif

// Async global->LDS copy of 16B per lane (tracked by ASYNCcnt) with a
// synchronous fallback if the builtin is unavailable on this toolchain.
DEV void cp_async_b128(unsigned short* l, const unsigned short* g) {
#if HAVE_ASYNC
  __builtin_amdgcn_global_load_async_to_lds_b128(
      (__attribute__((address_space(1))) int4v*)g,
      (__attribute__((address_space(3))) int4v*)l, 0, 0);
#else
  *(ushort8*)l = *(const ushort8*)g;
#endif
}

DEV void cp_async_wait() {
#if HAVE_ASYNC
#if __has_builtin(__builtin_amdgcn_s_wait_asynccnt)
  __builtin_amdgcn_s_wait_asynccnt(0);
#else
  asm volatile("s_wait_asynccnt 0x0" ::: "memory");
#endif
#endif
}

DEV unsigned short f2bf(float f) {
  union { float f; unsigned u; } x; x.f = f;
  unsigned r = x.u + 0x7FFFu + ((x.u >> 16) & 1u);   // round-to-nearest-even
  return (unsigned short)(r >> 16);
}

// Load one 16x32 bf16 WMMA operand fragment (A layout; B of a row-major "W"
// uses the identical pattern since column N of W^T is row N of W).
// p points at element [row_base = 0][k0]; ld in elements.
// ISA layout: lanes 0-15 hold K in [0..7] and [16..23]; lanes 16-31 hold
// K in [8..15] and [24..31]; lane's row = lane % 16; 2 values per VGPR.
DEV bf16x16 load_frag(const unsigned short* p, int ld) {
  int lane = threadIdx.x & 31;
  int m    = lane & 15;
  int g    = (lane >> 4) << 3;          // 0 or 8
  const unsigned short* r = p + m * ld + g;
  ushort8 lo = *(const ushort8*)(r);        // K = k0 + g + 0..7
  ushort8 hh = *(const ushort8*)(r + 16);   // K = k0 + 16 + g + 0..7
  FragCvt c;
#pragma unroll
  for (int i = 0; i < 8; ++i) { c.u[i] = lo[i]; c.u[i + 8] = hh[i]; }
  return c.v;
}

DEV floatx8 wmma_bf16(bf16x16 a, bf16x16 b, floatx8 c) {
  return __builtin_amdgcn_wmma_f32_16x16x32_bf16(
      /*neg_a=*/false, a, /*neg_b=*/false, b,
      /*c_mod=*/(short)0, c, /*reuse_a=*/false, /*reuse_b=*/false);
}

// ---------------------------------------------------------------------------
// fp32 -> bf16 conversion (4 elements / thread)
// ---------------------------------------------------------------------------
__global__ __launch_bounds__(256) void cvt_bf16x4(const float* __restrict__ in,
                                                  unsigned short* __restrict__ out,
                                                  int n4) {
  int i = blockIdx.x * 256 + threadIdx.x;
  if (i >= n4) return;
  float4 v = ((const float4*)in)[i];
  US4 o; o.a = f2bf(v.x); o.b = f2bf(v.y); o.c = f2bf(v.z); o.d = f2bf(v.w);
  ((US4*)out)[i] = o;
}

// cached K/V fp32 [B,3584,2048] -> bf16 rows of [B,4096,2048] (offset 0)
__global__ __launch_bounds__(256) void cvt_cache(const float* __restrict__ in,
                                                 unsigned short* __restrict__ out,
                                                 int n4) {
  int i = blockIdx.x * 256 + threadIdx.x;
  if (i >= n4) return;
  const int PB_IN  = 3584 * 2048 / 4;
  const int PB_OUT = 4096 * 2048 / 4;
  int b = i / PB_IN, rem = i - b * PB_IN;
  float4 v = ((const float4*)in)[i];
  US4 o; o.a = f2bf(v.x); o.b = f2bf(v.y); o.c = f2bf(v.z); o.d = f2bf(v.w);
  ((US4*)out)[(size_t)b * PB_OUT + rem] = o;
}

// ---------------------------------------------------------------------------
// GEMM: Y = X * W^T + bias.  X:[M,K] bf16 row-major, W:[N,K] bf16 row-major.
// Double-buffered 128x128 tile, K-step 64, async global->LDS staging.
// Output row mapping: orow = (m/rpb)*obs + oro + (m%rpb)   (KV-cache append)
// ---------------------------------------------------------------------------
template <bool BF16_OUT>
__global__ __launch_bounds__(256)
void gemm_xwt(const unsigned short* __restrict__ X,
              const unsigned short* __restrict__ W,
              const float* __restrict__ bias,
              void* __restrict__ Yp,
              int M, int N, int K,
              int rpb, int obs, int oro, int old) {
  __shared__ unsigned short Abuf[2][128 * 64];
  __shared__ unsigned short Bbuf[2][128 * 64];
  int n0 = blockIdx.x * 128, m0 = blockIdx.y * 128;
  int tid = threadIdx.x, wv = tid >> 5, lane = tid & 31;

  auto issue = [&](int k0, int s) {
#pragma unroll
    for (int rep = 0; rep < 4; ++rep) {
      int idx = tid + rep * 256;               // 1024 chunks of 8 bf16
      int r = idx >> 3, c = (idx & 7) << 3;    // 128 rows x 8 chunks
      cp_async_b128(&Abuf[s][r * 64 + c], &X[(size_t)(m0 + r) * K + k0 + c]);
      cp_async_b128(&Bbuf[s][r * 64 + c], &W[(size_t)(n0 + r) * K + k0 + c]);
    }
  };

  floatx8 acc[8];
#pragma unroll
  for (int i = 0; i < 8; ++i)
#pragma unroll
    for (int j = 0; j < 8; ++j) acc[i][j] = 0.0f;

  issue(0, 0);
  int p = 0;
  for (int k0 = 0; k0 < K; k0 += 64) {
    cp_async_wait();
    __syncthreads();                 // stage p fully populated, p^1 free
    if (k0 + 64 < K) issue(k0 + 64, p ^ 1);
#pragma unroll
    for (int kk = 0; kk < 2; ++kk) {
      bf16x16 afrag = load_frag(&Abuf[p][(wv * 16) * 64 + kk * 32], 64);
#pragma unroll
      for (int nt = 0; nt < 8; ++nt) {
        bf16x16 bfrag = load_frag(&Bbuf[p][(nt * 16) * 64 + kk * 32], 64);
        acc[nt] = wmma_bf16(afrag, bfrag, acc[nt]);
      }
    }
    p ^= 1;
  }

  int mrow = (lane >> 4) << 3;   // 0 or 8
  int ncol = lane & 15;
#pragma unroll
  for (int nt = 0; nt < 8; ++nt) {
    int n = n0 + nt * 16 + ncol;
    float bval = bias[n];
#pragma unroll
    for (int i = 0; i < 8; ++i) {
      int m = m0 + wv * 16 + i + mrow;
      int orow = (m / rpb) * obs + oro + (m % rpb);
      float v = acc[nt][i] + bval;
      if (BF16_OUT) ((unsigned short*)Yp)[(size_t)orow * old + n] = f2bf(v);
      else          ((float*)Yp)[(size_t)orow * old + n] = v;
    }
  }
}

// ---------------------------------------------------------------------------
// Flash attention.  Grid: (B*H, QLEN/128).  Block: 256 (8 waves x 16 q-rows).
// Software-pipelined: async K-chunk copy + V register loads for chunk t+32
// overlap with S/softmax/P.V compute on chunk t.  One block barrier / chunk.
// Qb:[B*512,2048] bf16.  Kt/Vt:[B*4096,2048] bf16.  Xo:[B*512,2048] bf16.
// ---------------------------------------------------------------------------
__global__ __launch_bounds__(256)
void attn_flash(const unsigned short* __restrict__ Qb,
                const unsigned short* __restrict__ Kt,
                const unsigned short* __restrict__ Vt,
                unsigned short* __restrict__ Xo) {
  const int CACHE = 3584, T = 4096;
  const float scale = 0.08838834764831845f;   // 1/sqrt(128)

  __shared__ unsigned short kb[2][32 * 128];  // [t][d]
  __shared__ unsigned short vb[2][128 * 32];  // V^T: [d][t]
  __shared__ unsigned short pbuf[8][16 * 32]; // per-wave P re-layout

  int b = blockIdx.x >> 4, h = blockIdx.x & 15;
  int wg_q0 = blockIdx.y * 128;
  int tid = threadIdx.x, wv = tid >> 5, lane = tid & 31;
  int q_base = wg_q0 + wv * 16;
  int mrow = (lane >> 4) << 3;   // 0 or 8
  int ncol = lane & 15;

  // cooperative-load indexing: 32 rows x 16 chunks of 8 bf16, 2 reps
  int cr0 = tid >> 4, cc0 = (tid & 15) << 3;          // rep 0
  int cr1 = (tid + 256) >> 4, cc1 = cc0;              // rep 1

  auto issue_k = [&](int t0, int s) {
    cp_async_b128(&kb[s][cr0 * 128 + cc0],
                  &Kt[(size_t)(b * 4096 + t0 + cr0) * 2048 + h * 128 + cc0]);
    cp_async_b128(&kb[s][cr1 * 128 + cc1],
                  &Kt[(size_t)(b * 4096 + t0 + cr1) * 2048 + h * 128 + cc1]);
  };
  auto load_v = [&](int t0, ushort8* vr) {
    vr[0] = *(const ushort8*)&Vt[(size_t)(b * 4096 + t0 + cr0) * 2048 + h * 128 + cc0];
    vr[1] = *(const ushort8*)&Vt[(size_t)(b * 4096 + t0 + cr1) * 2048 + h * 128 + cc1];
  };
  auto store_vT = [&](const ushort8* vr, int s) {
#pragma unroll
    for (int j = 0; j < 8; ++j) vb[s][(cc0 + j) * 32 + cr0] = vr[0][j];
#pragma unroll
    for (int j = 0; j < 8; ++j) vb[s][(cc1 + j) * 32 + cr1] = vr[1][j];
  };

  // Q fragments for this wave's 16 rows x 128 dims live in registers.
  const unsigned short* qptr = Qb + (size_t)(b * 512 + q_base) * 2048 + h * 128;
  bf16x16 qfrag[4];
#pragma unroll
  for (int kk = 0; kk < 4; ++kk) qfrag[kk] = load_frag(qptr + kk * 32, 2048);

  floatx8 acc[8];
  float row_max[8], row_sum[8];
#pragma unroll
  for (int i = 0; i < 8; ++i) {
    row_max[i] = -__builtin_inff();
    row_sum[i] = 0.0f;
#pragma unroll
    for (int j = 0; j < 8; ++j) acc[i][j] = 0.0f;
  }

  int t_hi = CACHE + wg_q0 + 128; if (t_hi > T) t_hi = T;

  // ---- pipeline prologue: stage chunk 0 ----
  {
    ushort8 vr[2];
    issue_k(0, 0);
    load_v(0, vr);
    cp_async_wait();
    store_vT(vr, 0);
  }
  __syncthreads();

  int p = 0;
  for (int t0 = 0; t0 < t_hi; t0 += 32) {
    int tn = t0 + 32;
    bool more = tn < t_hi;
    ushort8 vr[2];
    if (more) { issue_k(tn, p ^ 1); load_v(tn, vr); }   // overlap with compute

    // S = Q * K^T  (16 x 32 tile, two 16x16 WMMA accumulators over d=128)
    floatx8 s0, s1;
#pragma unroll
    for (int j = 0; j < 8; ++j) { s0[j] = 0.0f; s1[j] = 0.0f; }
#pragma unroll
    for (int kk = 0; kk < 4; ++kk) {
      bf16x16 k0f = load_frag(&kb[p][kk * 32], 128);
      bf16x16 k1f = load_frag(&kb[p][16 * 128 + kk * 32], 128);
      s0 = wmma_bf16(qfrag[kk], k0f, s0);
      s1 = wmma_bf16(qfrag[kk], k1f, s1);
    }

    // Online softmax in C-layout (row = i + mrow, col = ncol [+16]).
    float corr[8];
#pragma unroll
    for (int i = 0; i < 8; ++i) {
      float a0 = s0[i] * scale, a1 = s1[i] * scale;
      int lim = q_base + i + mrow + CACHE;     // attend to t <= lim
      if (t0 + ncol > lim)      a0 = -__builtin_inff();
      if (t0 + 16 + ncol > lim) a1 = -__builtin_inff();
      float mx = fmaxf(a0, a1);
#pragma unroll
      for (int d = 1; d < 16; d <<= 1) mx = fmaxf(mx, __shfl_xor(mx, d, 32));
      float nm = fmaxf(row_max[i], mx);
      float cr = __expf(row_max[i] - nm);
      row_max[i] = nm;
      float p0 = __expf(a0 - nm), p1 = __expf(a1 - nm);
      float sm = p0 + p1;
#pragma unroll
      for (int d = 1; d < 16; d <<= 1) sm += __shfl_xor(sm, d, 32);
      row_sum[i] = row_sum[i] * cr + sm;
      corr[i] = cr;
      int mr = i + mrow;
      pbuf[wv][mr * 32 + ncol]      = f2bf(p0);
      pbuf[wv][mr * 32 + 16 + ncol] = f2bf(p1);
    }
    // pbuf is wave-private and LDS is in-order per wave; a wave-level
    // scheduling barrier is enough to order the cross-lane write->read.
    __builtin_amdgcn_wave_barrier();

#pragma unroll
    for (int nt = 0; nt < 8; ++nt)
#pragma unroll
      for (int i = 0; i < 8; ++i) acc[nt][i] *= corr[i];

    // O += P * V   (B operand = rows of V^T in LDS)
    bf16x16 pfrag = load_frag(pbuf[wv], 32);
#pragma unroll
    for (int nt = 0; nt < 8; ++nt) {
      bf16x16 vfrag = load_frag(&vb[p][(nt * 16) * 32], 32);
      acc[nt] = wmma_bf16(pfrag, vfrag, acc[nt]);
    }

    // ---- drain pipeline for next chunk ----
    cp_async_wait();                 // kb[p^1] landed
    if (more) store_vT(vr, p ^ 1);   // vb[p^1] free since last barrier
    __syncthreads();
    p ^= 1;
  }

#pragma unroll
  for (int i = 0; i < 8; ++i) row_sum[i] = 1.0f / row_sum[i];
#pragma unroll
  for (int nt = 0; nt < 8; ++nt)
#pragma unroll
    for (int i = 0; i < 8; ++i) {
      size_t row = (size_t)(b * 512 + q_base + i + mrow);
      int col = h * 128 + nt * 16 + ncol;
      Xo[row * 2048 + col] = f2bf(acc[nt][i] * row_sum[i]);
    }
}

// ---------------------------------------------------------------------------
extern "C" void kernel_launch(void* const* d_in, const int* in_sizes, int n_in,
                              void* d_out, int out_size, void* d_ws, size_t ws_size,
                              hipStream_t stream) {
  (void)in_sizes; (void)n_in; (void)out_size; (void)ws_size;
  const float* query    = (const float*)d_in[0];
  const float* key      = (const float*)d_in[1];
  const float* value    = (const float*)d_in[2];
  const float* cached_k = (const float*)d_in[3];
  const float* cached_v = (const float*)d_in[4];
  const float* wq = (const float*)d_in[5];  const float* bq = (const float*)d_in[6];
  const float* wk = (const float*)d_in[7];  const float* bk = (const float*)d_in[8];
  const float* wv = (const float*)d_in[9];  const float* bv = (const float*)d_in[10];
  const float* wo = (const float*)d_in[11]; const float* bo = (const float*)d_in[12];

  const size_t NTOK = 4 * 512;        // 2048 query tokens
  const size_t D    = 2048;
  const size_t KVROWS = (size_t)4 * 4096;

  char* ws = (char*)d_ws;
  size_t off = 0;
  auto carve = [&](size_t elems) -> unsigned short* {
    unsigned short* p = (unsigned short*)(ws + off);
    off = (off + elems * 2 + 255) & ~(size_t)255;
    return p;
  };
  unsigned short* qin  = carve(NTOK * D);
  unsigned short* kin  = carve(NTOK * D);
  unsigned short* vin  = carve(NTOK * D);
  unsigned short* wqb  = carve(D * D);
  unsigned short* wkb  = carve(D * D);
  unsigned short* wvb  = carve(D * D);
  unsigned short* wob  = carve(D * D);
  unsigned short* Qb   = carve(NTOK * D);
  unsigned short* Ktot = carve(KVROWS * D);
  unsigned short* Vtot = carve(KVROWS * D);
  unsigned short* Xat  = carve(NTOK * D);

  // --- convert inputs / weights to bf16 ---
  {
    int n4 = (int)(NTOK * D / 4);                 // 1,048,576
    int g  = (n4 + 255) / 256;
    cvt_bf16x4<<<g, 256, 0, stream>>>(query, qin, n4);
    cvt_bf16x4<<<g, 256, 0, stream>>>(key,   kin, n4);
    cvt_bf16x4<<<g, 256, 0, stream>>>(value, vin, n4);
    int w4 = (int)(D * D / 4);
    int gw = (w4 + 255) / 256;
    cvt_bf16x4<<<gw, 256, 0, stream>>>(wq, wqb, w4);
    cvt_bf16x4<<<gw, 256, 0, stream>>>(wk, wkb, w4);
    cvt_bf16x4<<<gw, 256, 0, stream>>>(wv, wvb, w4);
    cvt_bf16x4<<<gw, 256, 0, stream>>>(wo, wob, w4);
    int c4 = (int)((size_t)4 * 3584 * 2048 / 4);  // 7,340,032
    int gc = (c4 + 255) / 256;
    cvt_cache<<<gc, 256, 0, stream>>>(cached_k, Ktot, c4);
    cvt_cache<<<gc, 256, 0, stream>>>(cached_v, Vtot, c4);
  }

  // --- projections (WMMA GEMMs) ---
  dim3 gg(16, 16);   // N/128, M/128
  gemm_xwt<true><<<gg, 256, 0, stream>>>(qin, wqb, bq, Qb,
      2048, 2048, 2048, /*rpb=*/512, /*obs=*/512,  /*oro=*/0,    /*old=*/2048);
  gemm_xwt<true><<<gg, 256, 0, stream>>>(kin, wkb, bk, Ktot,
      2048, 2048, 2048, /*rpb=*/512, /*obs=*/4096, /*oro=*/3584, /*old=*/2048);
  gemm_xwt<true><<<gg, 256, 0, stream>>>(vin, wvb, bv, Vtot,
      2048, 2048, 2048, /*rpb=*/512, /*obs=*/4096, /*oro=*/3584, /*old=*/2048);

  // --- flash attention ---
  attn_flash<<<dim3(64, 4), 256, 0, stream>>>(Qb, Ktot, Vtot, Xat);

  // --- output projection (fp32 out) ---
  gemm_xwt<false><<<gg, 256, 0, stream>>>(Xat, wob, bo, d_out,
      2048, 2048, 2048, /*rpb=*/512, /*obs=*/512, /*oro=*/0, /*old=*/2048);
}
